// TransformerBlock_63539746177335
// MI455X (gfx1250) — matrix operand order repaired and checked by reading, not compiled, not run on hardware
//
#include <hip/hip_runtime.h>

typedef __bf16 bf16;
typedef __attribute__((ext_vector_type(16))) __bf16 v16bf;
typedef __attribute__((ext_vector_type(8)))  float  v8f;

typedef unsigned int u32x4 __attribute__((ext_vector_type(4)));
typedef int          i32x4 __attribute__((ext_vector_type(4)));
typedef int          i32x8 __attribute__((ext_vector_type(8)));

constexpr int CB     = 2;
constexpr int CSEQ   = 2048;
constexpr int CDIM   = 1024;
constexpr int CHEADS = 16;
constexpr int CHD    = 64;
constexpr int CHID   = 2816;
constexpr float CEPS   = 1e-5f;
constexpr float CLNEPS = 1e-5f;

#if __has_builtin(__builtin_amdgcn_tensor_load_to_lds)
#define HAVE_TDM 1
#else
#define HAVE_TDM 0
#endif

#if __has_builtin(__builtin_amdgcn_s_wait_tensorcnt)
#define WAIT_TENSORCNT0() __builtin_amdgcn_s_wait_tensorcnt((short)0)
#else
#define WAIT_TENSORCNT0() ((void)0)
#endif

// ---------------------------------------------------------------- WMMA utils

__device__ inline v8f wmma_bf(v16bf a, v16bf b, v8f c) {
  // v_wmma_f32_16x16x32_bf16  D = A*B + C
  return __builtin_amdgcn_wmma_f32_16x16x32_bf16(false, a, false, b,
                                                 (short)0, c, false, false);
}

// A fragment (16x32 bf16, MxK): lane holds row m = lane&15; kh = lane>>4.
// elements e <- row[kh*16 + e]
__device__ inline v16bf load_a_frag(const bf16* row, int lane) {
  int kh = (lane >> 4) & 1;
  v16bf a;
  ((uint4*)&a)[0] = *(const uint4*)(row + kh * 16);
  ((uint4*)&a)[1] = *(const uint4*)(row + kh * 16 + 8);
  return a;
}

// B fragment (32x16 bf16, KxN): lane holds col n = lane&15; kh = lane>>4.
// elements e <- col[kh*16 + e]   (colbase = 32 contiguous K values for col n)
__device__ inline v16bf load_b_frag(const bf16* colbase, int lane) {
  int kh = (lane >> 4) & 1;
  v16bf b;
  ((uint4*)&b)[0] = *(const uint4*)(colbase + kh * 16);
  ((uint4*)&b)[1] = *(const uint4*)(colbase + kh * 16 + 8);
  return b;
}

// ---------------------------------------------------------------- TDM

#if HAVE_TDM
// DMA a 2D bf16 tile (tileRows x tileCols) from a row-major tensor into LDS,
// with hardware row padding per the D# pad fields.  Issued once per calling
// wave; completion tracked by TENSORcnt.
__device__ inline void tdm_load_tile_2d_bf16(const bf16* g, unsigned ldsOff,
                                             unsigned tileCols, unsigned tileRows,
                                             unsigned tensorCols, unsigned tensorRows,
                                             unsigned long long rowPitchElems,
                                             unsigned padIntCode, unsigned padAmtCode) {
  unsigned long long ga = (unsigned long long)g;
  u32x4 g0;
  g0[0] = 1u;                                                  // count=1 (valid)
  g0[1] = ldsOff;                                              // lds_addr (bytes)
  g0[2] = (unsigned)ga;                                        // global_addr lo
  g0[3] = (unsigned)((ga >> 32) & 0x01ffffffu) | (2u << 30);   // addr hi | type=2
  i32x8 g1;
  g1[0] = (int)((1u << 16)                 // data_size = 1 (2 bytes)
              | (1u << 20)                 // pad_enable
              | (padIntCode << 22)         // pad interval (2^(c+1) DWORDs)
              | (padAmtCode << 25));       // pad amount   (c+1 DWORDs)
  g1[1] = (int)((tensorCols & 0xffffu) << 16);                 // dim0[15:0]
  g1[2] = (int)(((tensorCols >> 16) & 0xffffu) |
                ((tensorRows & 0xffffu) << 16));               // dim0[31:16]|dim1
  g1[3] = (int)(((tensorRows >> 16) & 0xffffu) |
                ((tileCols & 0xffffu) << 16));                 // dim1[31:16]|tile0
  g1[4] = (int)(tileRows & 0xffffu);                           // tile1; tile2=0
  g1[5] = (int)(unsigned)(rowPitchElems & 0xffffffffu);        // dim0_stride lo
  g1[6] = (int)(unsigned)((rowPitchElems >> 32) & 0xffffu);    // stride hi; s1=0
  g1[7] = 0;
  i32x4 zz = {};
#if defined(__clang_major__) && (__clang_major__ >= 23)
  i32x8 z8 = {};
  __builtin_amdgcn_tensor_load_to_lds(g0, g1, zz, zz, z8, 0);
#else
  __builtin_amdgcn_tensor_load_to_lds(g0, g1, zz, zz, 0);
#endif
}
#endif  // HAVE_TDM

// ---------------------------------------------------------------- reductions

__device__ inline float block_sum(float val, float* red) {
#pragma unroll
  for (int d = 16; d > 0; d >>= 1) val += __shfl_xor(val, d, 32);
  int wid = threadIdx.x >> 5, lane = threadIdx.x & 31;
  if (lane == 0) red[wid] = val;
  __syncthreads();
  if (wid == 0) {
    float r = (lane < 8) ? red[lane] : 0.f;
#pragma unroll
    for (int d = 4; d > 0; d >>= 1) r += __shfl_xor(r, d, 32);
    if (lane == 0) red[0] = r;
  }
  __syncthreads();
  float r = red[0];
  __syncthreads();
  return r;
}

// ---------------------------------------------------------------- small ops

__global__ void f2bf_kernel(const float* __restrict__ s, bf16* __restrict__ d,
                            size_t n) {
  size_t i = (size_t)blockIdx.x * blockDim.x + threadIdx.x;
  if (i < n) d[i] = (bf16)s[i];
}

__global__ __launch_bounds__(256)
void adaln_kernel(const float* __restrict__ ada, const float* __restrict__ aw,
                  const float* __restrict__ ab, float* __restrict__ m) {
  __shared__ float sa[CDIM];
  int b = blockIdx.y;
  int j = blockIdx.x * 256 + threadIdx.x;          // 0..6*CDIM-1
  for (int i = threadIdx.x; i < CDIM; i += 256) {
    float x = ada[(size_t)b * CDIM + i];
    sa[i] = x / (1.f + __expf(-x));                // silu
  }
  __syncthreads();
  float acc = ab[j];
  for (int kc = 0; kc < CDIM; ++kc)
    acc += sa[kc] * aw[(size_t)kc * (6 * CDIM) + j];
  m[(size_t)b * 6 * CDIM + j] = acc;
}

__global__ __launch_bounds__(256)
void rms_mod_kernel(const float* __restrict__ x, const float* __restrict__ w,
                    const float* __restrict__ m, int shift_off, int scale_off,
                    bf16* __restrict__ out) {
  __shared__ float red[8];
  int row = blockIdx.x;
  int b = row / CSEQ;
  const float* xr = x + (size_t)row * CDIM;
  const float* sh = m + (size_t)b * 6 * CDIM + shift_off;
  const float* sc = m + (size_t)b * 6 * CDIM + scale_off;
  float ss = 0.f;
  for (int i = threadIdx.x; i < CDIM; i += 256) { float t = xr[i]; ss += t * t; }
  ss = block_sum(ss, red);
  float inv = rsqrtf(ss / CDIM + CEPS);
  for (int i = threadIdx.x; i < CDIM; i += 256)
    out[(size_t)row * CDIM + i] =
        (bf16)(xr[i] * inv * w[i] * (1.f + sc[i]) + sh[i]);
}

__global__ __launch_bounds__(256)
void ln_rope_kernel(const float* __restrict__ x, const float* __restrict__ w,
                    const float* __restrict__ bb, const float* __restrict__ fc,
                    const float* __restrict__ fs, bf16* __restrict__ out) {
  __shared__ float red[8];
  int row = blockIdx.x;
  int pos = row % CSEQ;
  const float* xr = x + (size_t)row * CDIM;
  float s1 = 0.f, s2 = 0.f;
  for (int i = threadIdx.x; i < CDIM; i += 256) { float t = xr[i]; s1 += t; s2 += t * t; }
  s1 = block_sum(s1, red);
  s2 = block_sum(s2, red);
  float mu  = s1 / CDIM;
  float var = s2 / CDIM - mu * mu;
  float inv = rsqrtf(var + CLNEPS);
  for (int p = threadIdx.x; p < CDIM / 2; p += 256) {
    int d0 = 2 * p, d1 = 2 * p + 1;
    float t0 = (xr[d0] - mu) * inv * w[d0] + bb[d0];
    float t1 = (xr[d1] - mu) * inv * w[d1] + bb[d1];
    int i = p & (CHD / 2 - 1);                       // pair index within head
    float c = fc[(size_t)pos * (CHD / 2) + i];
    float s = fs[(size_t)pos * (CHD / 2) + i];
    out[(size_t)row * CDIM + d0] = (bf16)(t0 * c - t1 * s);
    out[(size_t)row * CDIM + d1] = (bf16)(t0 * s + t1 * c);
  }
}

__global__ void resgate_kernel(const float* __restrict__ x,
                               const float* __restrict__ y,
                               const float* __restrict__ m, int gate_off,
                               float* __restrict__ out, size_t n) {
  size_t i = (size_t)blockIdx.x * blockDim.x + threadIdx.x;
  if (i >= n) return;
  int b = (int)(i / ((size_t)CSEQ * CDIM));
  int c = (int)(i % CDIM);
  out[i] = x[i] + m[(size_t)b * 6 * CDIM + gate_off + c] * y[i];
}

__global__ void silumul_kernel(const bf16* __restrict__ g1,
                               const bf16* __restrict__ g3, bf16* gg, size_t n) {
  size_t i = (size_t)blockIdx.x * blockDim.x + threadIdx.x;
  if (i < n) {
    float a = (float)g1[i], c = (float)g3[i];
    gg[i] = (bf16)((a / (1.f + __expf(-a))) * c);
  }
}

// ---------------------------------------------------------------- WMMA GEMM
// C[M,N] = A[M,K] * B[K,N]; A,B bf16 row-major; 8 waves, each owns 32x32.
// Double-buffered software pipeline: the Tensor Data Mover DMAs the NEXT
// 128x32 A tile (with hardware row padding: 64B data + 16B pad = 80B stride
// == As[][BK+8]) while the WMMA pipe computes on the CURRENT tile; B tiles
// are transposed cooperatively into the inactive buffer.  One barrier per
// iteration; TDM completion via TENSORcnt.

template <int STORE_BF16>
__global__ __launch_bounds__(256)
void gemm_bf16_wmma(const bf16* __restrict__ A, const bf16* __restrict__ Bm,
                    void* __restrict__ Cp, int M, int N, int K) {
  constexpr int BMt = 128, BNt = 64, BKt = 32, PAD = 8;
  __shared__ bf16 As[2][BMt][BKt + PAD];   // 2 x 10240B, row stride 80B
  __shared__ bf16 Bs[2][BNt][BKt + PAD];   // 2 x  5120B, transposed: [n][k]
  const int tid = threadIdx.x, lane = tid & 31, wid = tid >> 5;
  const int bm = blockIdx.x * BMt, bn = blockIdx.y * BNt;
  const int wm = (wid & 3) * 32, wn = (wid >> 2) * 32;

  auto loadB = [&](int k0, int bufi) {
    int kr = tid >> 3, nc = (tid & 7) * 8;
    uint4 t = *(const uint4*)(Bm + (size_t)(k0 + kr) * N + bn + nc);
    const bf16* tb = (const bf16*)&t;
#pragma unroll
    for (int j = 0; j < 8; ++j) Bs[bufi][nc + j][kr] = tb[j];
  };
#if HAVE_TDM
  const unsigned asLds0 = (unsigned)(unsigned long long)(uintptr_t)&As[0][0][0];
  const unsigned asLds1 = (unsigned)(unsigned long long)(uintptr_t)&As[1][0][0];
  auto loadA = [&](int k0, int bufi) {
    if (wid == 0)
      tdm_load_tile_2d_bf16(A + (size_t)bm * K + k0, bufi ? asLds1 : asLds0,
                            (unsigned)BKt, (unsigned)BMt,
                            (unsigned)K, (unsigned)M,
                            (unsigned long long)K,
                            /*padIntCode 16DW=*/3u, /*padAmtCode 4DW=*/3u);
  };
#else
  auto loadA = [&](int k0, int bufi) {
#pragma unroll
    for (int it = 0; it < 2; ++it) {
      int ch = tid + it * 256;
      int r = ch >> 2, c = (ch & 3) * 8;
      *(uint4*)&As[bufi][r][c] = *(const uint4*)(A + (size_t)(bm + r) * K + k0 + c);
    }
  };
#endif

  v8f acc[2][2] = {};
  int buf = 0;
  loadA(0, 0);                        // prologue: tile 0 in flight
  loadB(0, 0);
  for (int k0 = 0; k0 < K; k0 += BKt) {
#if HAVE_TDM
    if (wid == 0) WAIT_TENSORCNT0();  // TDM for `buf` done (overlapped w/ prev compute)
#endif
    __syncthreads();                  // publish A(TDM) + B(coop) for `buf`
    int nxt = buf ^ 1;
    if (k0 + BKt < K) {               // issue NEXT tile; overlaps compute below
      loadA(k0 + BKt, nxt);
      loadB(k0 + BKt, nxt);
      __builtin_prefetch(Bm + (size_t)(k0 + 2 * BKt + (tid >> 3)) * N + bn, 0, 3);
    }
    v16bf af[2], bfr[2];
#pragma unroll
    for (int t = 0; t < 2; ++t)
      af[t] = load_a_frag(&As[buf][wm + t * 16 + (lane & 15)][0], lane);
#pragma unroll
    for (int t = 0; t < 2; ++t)
      bfr[t] = load_b_frag(&Bs[buf][wn + t * 16 + (lane & 15)][0], lane);
#pragma unroll
    for (int tm = 0; tm < 2; ++tm)
#pragma unroll
      for (int tn = 0; tn < 2; ++tn)
        acc[tm][tn] = wmma_bf(af[tm], bfr[tn], acc[tm][tn]);
    buf = nxt;
  }
  const int kh = lane >> 4, col = lane & 15;
#pragma unroll
  for (int tm = 0; tm < 2; ++tm)
#pragma unroll
    for (int tn = 0; tn < 2; ++tn)
#pragma unroll
      for (int g = 0; g < 8; ++g) {
        int mg = bm + wm + tm * 16 + g + 8 * kh;   // C layout: M = g + 8*kh
        int ng = bn + wn + tn * 16 + col;          //           N = lane&15
        float v = acc[tm][tn][g];
        if (STORE_BF16) ((bf16*)Cp)[(size_t)mg * N + ng] = (bf16)v;
        else            ((float*)Cp)[(size_t)mg * N + ng] = v;
      }
}

// ---------------------------------------------------------------- attention
// Flash-style: one wave per 16-row Q tile, loop keys in 32-tiles.
// q,k,v,o: (B, SEQ, HEADS, HD) bf16.

__global__ __launch_bounds__(256)
void attn_fwd(const bf16* __restrict__ q, const bf16* __restrict__ kk,
              const bf16* __restrict__ v, bf16* __restrict__ o, float scale) {
  const int b = blockIdx.z, h = blockIdx.y;
  const int lane = threadIdx.x & 31, wid = threadIdx.x >> 5;
  const int qbase = blockIdx.x * 128 + wid * 16;
  __shared__ bf16 plds[8][16][32];                 // per-wave P tile (bf16)
  const size_t rs = (size_t)CHEADS * CHD;          // = CDIM
  const bf16* qp = q  + ((size_t)b * CSEQ) * rs + h * CHD;
  const bf16* kp = kk + ((size_t)b * CSEQ) * rs + h * CHD;
  const bf16* vp = v  + ((size_t)b * CSEQ) * rs + h * CHD;
  const int kh = lane >> 4, col = lane & 15;
  const bf16* qrow = qp + (size_t)(qbase + col) * rs;
  v16bf qf0 = load_a_frag(qrow, lane);             // d = 0..31
  v16bf qf1 = load_a_frag(qrow + 32, lane);        // d = 32..63
  v8f oacc[4] = {};
  float mrun[8], lrun[8];
#pragma unroll
  for (int g = 0; g < 8; ++g) { mrun[g] = -1e30f; lrun[g] = 0.f; }

  for (int kt = 0; kt < CSEQ; kt += 32) {
    // S = Q * K^T : two 16-key halves, reduction over d (2 wmma each)
    v8f s[2];
#pragma unroll
    for (int half = 0; half < 2; ++half) {
      const bf16* krow = kp + (size_t)(kt + half * 16 + col) * rs;
      v8f sh = {};
      sh = wmma_bf(qf0, load_b_frag(krow, lane), sh);
      sh = wmma_bf(qf1, load_b_frag(krow + 32, lane), sh);
      s[half] = sh;
    }
    // online softmax: C-layout row = g + 8*kh, stats over 16-lane groups
#pragma unroll
    for (int g = 0; g < 8; ++g) {
      float v0 = s[0][g] * scale;
      float v1 = s[1][g] * scale;
      float mx = fmaxf(v0, v1);
#pragma unroll
      for (int d = 1; d < 16; d <<= 1) mx = fmaxf(mx, __shfl_xor(mx, d, 32));
      float mn = fmaxf(mrun[g], mx);
      float corr = __expf(mrun[g] - mn);
      lrun[g] *= corr;
#pragma unroll
      for (int dt = 0; dt < 4; ++dt) oacc[dt][g] *= corr;
      float p0 = __expf(v0 - mn);
      float p1 = __expf(v1 - mn);
      float ps = p0 + p1;
#pragma unroll
      for (int d = 1; d < 16; d <<= 1) ps += __shfl_xor(ps, d, 32);
      lrun[g] += ps;
      mrun[g] = mn;
      int m = g + 8 * kh;
      plds[wid][m][col]      = (bf16)p0;           // keys kt..kt+15
      plds[wid][m][16 + col] = (bf16)p1;           // keys kt+16..kt+31
    }
    asm volatile("s_wait_dscnt 0" ::: "memory");   // P write -> P read (same wave)
    // O += P * V : P as A-frag (16x32), V tiles as B-frags (32x16 per d-tile)
    v16bf pf = load_a_frag(&plds[wid][col][0], lane);
#pragma unroll
    for (int dt = 0; dt < 4; ++dt) {
      const bf16* vb = vp + (size_t)(kt + kh * 16) * rs + dt * 16 + col;
      v16bf vf;
#pragma unroll
      for (int e = 0; e < 16; ++e) vf[e] = vb[(size_t)e * rs];
      oacc[dt] = wmma_bf(pf, vf, oacc[dt]);
    }
  }
#pragma unroll
  for (int dt = 0; dt < 4; ++dt)
#pragma unroll
    for (int g = 0; g < 8; ++g) {
      int row = qbase + g + 8 * kh;
      o[((size_t)(b * CSEQ + row)) * rs + h * CHD + dt * 16 + col] =
          (bf16)(oacc[dt][g] / lrun[g]);
    }
}

// ---------------------------------------------------------------- launch

extern "C" void kernel_launch(void* const* d_in, const int* in_sizes, int n_in,
                              void* d_out, int out_size, void* d_ws,
                              size_t ws_size, hipStream_t stream) {
  const float* x    = (const float*)d_in[0];
  const float* ada  = (const float*)d_in[1];
  const float* fcos = (const float*)d_in[2];
  const float* fsin = (const float*)d_in[3];
  const float* wq   = (const float*)d_in[4];
  const float* wk   = (const float*)d_in[5];
  const float* wv   = (const float*)d_in[6];
  const float* wo   = (const float*)d_in[7];
  const float* qnw  = (const float*)d_in[8];
  const float* qnb  = (const float*)d_in[9];
  const float* knw  = (const float*)d_in[10];
  const float* knb  = (const float*)d_in[11];
  const float* anw  = (const float*)d_in[12];
  const float* fnw  = (const float*)d_in[13];
  const float* adaw = (const float*)d_in[14];
  const float* adab = (const float*)d_in[15];
  const float* w1   = (const float*)d_in[16];
  const float* w2   = (const float*)d_in[17];
  const float* w3   = (const float*)d_in[18];
  float* out = (float*)d_out;

  const int M   = CB * CSEQ;                 // 4096
  const size_t nTok = (size_t)M * CDIM;      // 4,194,304
  const size_t nHid = (size_t)M * CHID;      // 11,534,336

  char* ws = (char*)d_ws;
  size_t off = 0;
  auto alloc = [&](size_t bytes) -> void* {
    void* p = ws + off;
    off += (bytes + 255) & ~(size_t)255;
    return p;
  };

  bf16* wq_b = (bf16*)alloc((size_t)CDIM * CDIM * 2);
  bf16* wk_b = (bf16*)alloc((size_t)CDIM * CDIM * 2);
  bf16* wv_b = (bf16*)alloc((size_t)CDIM * CDIM * 2);
  bf16* wo_b = (bf16*)alloc((size_t)CDIM * CDIM * 2);
  bf16* w1_b = (bf16*)alloc((size_t)CDIM * CHID * 2);
  bf16* w3_b = (bf16*)alloc((size_t)CDIM * CHID * 2);
  bf16* w2_b = (bf16*)alloc((size_t)CHID * CDIM * 2);
  float* mbuf   = (float*)alloc((size_t)CB * 6 * CDIM * 4);
  bf16* h_b     = (bf16*)alloc(nTok * 2);          // h, reused as h2
  float* xq_f   = (float*)alloc(nTok * 4);         // xq, reused as attn_out
  float* xk_f   = (float*)alloc(nTok * 4);         // xk, reused as ffn_out
  bf16* q_b     = (bf16*)alloc(nTok * 2);
  bf16* k_b     = (bf16*)alloc(nTok * 2);
  bf16* v_b     = (bf16*)alloc(nTok * 2);
  bf16* attn_b  = (bf16*)alloc(nTok * 2);
  float* xmid_f = (float*)alloc(nTok * 4);
  bf16* g1_b    = (bf16*)alloc(nHid * 2);
  bf16* g3_b    = (bf16*)alloc(nHid * 2);

  auto cvt = [&](const float* s, bf16* d, size_t n) {
    f2bf_kernel<<<(int)((n + 255) / 256), 256, 0, stream>>>(s, d, n);
  };
  cvt(wq, wq_b, (size_t)CDIM * CDIM);
  cvt(wk, wk_b, (size_t)CDIM * CDIM);
  cvt(wv, wv_b, (size_t)CDIM * CDIM);
  cvt(wo, wo_b, (size_t)CDIM * CDIM);
  cvt(w1, w1_b, (size_t)CDIM * CHID);
  cvt(w3, w3_b, (size_t)CDIM * CHID);
  cvt(w2, w2_b, (size_t)CHID * CDIM);

  // adaLN modulation vector m = silu(ada) @ ada_w + ada_b
  adaln_kernel<<<dim3(6 * CDIM / 256, CB), 256, 0, stream>>>(ada, adaw, adab, mbuf);

  // h = modulate(rmsnorm(x), shift_msa, scale_msa)   [bf16]
  rms_mod_kernel<<<M, 256, 0, stream>>>(x, anw, mbuf, 0, CDIM, h_b);

  // QKV projections
  dim3 gD(M / 128, CDIM / 64);
  gemm_bf16_wmma<0><<<gD, 256, 0, stream>>>(h_b, wq_b, xq_f, M, CDIM, CDIM);
  gemm_bf16_wmma<0><<<gD, 256, 0, stream>>>(h_b, wk_b, xk_f, M, CDIM, CDIM);
  gemm_bf16_wmma<1><<<gD, 256, 0, stream>>>(h_b, wv_b, v_b,  M, CDIM, CDIM);

  // layernorm + RoPE -> bf16 q/k
  ln_rope_kernel<<<M, 256, 0, stream>>>(xq_f, qnw, qnb, fcos, fsin, q_b);
  ln_rope_kernel<<<M, 256, 0, stream>>>(xk_f, knw, knb, fcos, fsin, k_b);

  // flash attention
  attn_fwd<<<dim3(CSEQ / 128, CHEADS, CB), 256, 0, stream>>>(
      q_b, k_b, v_b, attn_b, 0.125f);

  // output projection + gated residual
  gemm_bf16_wmma<0><<<gD, 256, 0, stream>>>(attn_b, wo_b, xq_f, M, CDIM, CDIM);
  resgate_kernel<<<(int)((nTok + 255) / 256), 256, 0, stream>>>(
      x, xq_f, mbuf, 2 * CDIM, xmid_f, nTok);

  // FFN
  rms_mod_kernel<<<M, 256, 0, stream>>>(xmid_f, fnw, mbuf, 3 * CDIM, 4 * CDIM, h_b);
  dim3 gU(M / 128, CHID / 64);
  gemm_bf16_wmma<1><<<gU, 256, 0, stream>>>(h_b, w1_b, g1_b, M, CHID, CDIM);
  gemm_bf16_wmma<1><<<gU, 256, 0, stream>>>(h_b, w3_b, g3_b, M, CHID, CDIM);
  silumul_kernel<<<(int)((nHid + 255) / 256), 256, 0, stream>>>(g1_b, g3_b, g1_b, nHid);
  gemm_bf16_wmma<0><<<gD, 256, 0, stream>>>(g1_b, w2_b, xk_f, M, CDIM, CHID);
  resgate_kernel<<<(int)((nTok + 255) / 256), 256, 0, stream>>>(
      xmid_f, xk_f, mbuf, 5 * CDIM, out, nTok);
}
